// GraphAttentionConvolution_12378095747834
// MI455X (gfx1250) — matrix operand order repaired and checked by reading, not compiled, not run on hardware
//
#include <hip/hip_runtime.h>
#include <hip/hip_bf16.h>
#include <math.h>

typedef __attribute__((ext_vector_type(16))) __bf16 v16bf;
typedef __attribute__((ext_vector_type(8)))  float  v8f;

#define N_NODES 4096
#define DIN     1024
#define DOUT    64
#define GAMMA_F 0.1f

// Packed-W layout (bf16, WMMA B-fragment order):
//   Wpack[((mat*4 + ntile)*32 + kstep)*32 + lane][0..15]
// where for lane: half = lane>>4, n = lane&15, and element i holds
//   W[(kstep*32 + 16*half + i)*64 + ntile*16 + n]
// => each lane's 16 bf16 (32B) are contiguous: two b128 loads per fragment.

// ---------------------------------------------------------------------------
// Kernel 0: pack W1/W2 into bf16 WMMA B-fragment layout.
// One thread per (mat, ntile, kstep, lane) = 8192 threads.
// ---------------------------------------------------------------------------
__global__ __launch_bounds__(256) void gat_packw_kernel(
    const float* __restrict__ W1, const float* __restrict__ W2,
    __bf16* __restrict__ Wpack)
{
    const int t = blockIdx.x * blockDim.x + threadIdx.x;   // 0..8191
    if (t >= 2 * 4 * 32 * 32) return;
    const int lane  = t & 31;
    const int kstep = (t >> 5) & 31;
    const int nt    = (t >> 10) & 3;
    const int mat   = t >> 12;

    const float* W = mat ? W2 : W1;
    const int half = lane >> 4;
    const int n    = lane & 15;
    const int kb   = kstep * 32 + 16 * half;

    v16bf v;
#pragma unroll
    for (int i = 0; i < 16; ++i)
        v[i] = (__bf16)W[(size_t)(kb + i) * DOUT + nt * 16 + n];
    *(v16bf*)(Wpack + (size_t)t * 16) = v;
}

// ---------------------------------------------------------------------------
// Kernel 1: h = x @ W for (x1,W1)->h1 and (x2,W2)->h2 using bf16 WMMA.
// One wave per 16-row tile; each wave computes the full 16x64 strip
// (4 N-tiles) reusing the A fragment across 4 WMMAs per K-step.
// VGPR layouts used here (wave32):
//   A 16x32 bf16 : half h = lane>>4, row M = lane&15, elem i -> K = 16h + i
//   B 32x16 bf16 : col N = lane&15, elem i -> K = 16h + i   (pre-packed)
//   C/D f32 16x16: vgpr r -> M = r + 8h, N = lane&15
// ---------------------------------------------------------------------------
__global__ __launch_bounds__(256) void gat_gemm_h_kernel(
    const float* __restrict__ x1, const float* __restrict__ x2,
    const __bf16* __restrict__ Wpack,
    float* __restrict__ h1, float* __restrict__ h2)
{
    const int lane = threadIdx.x & 31;
    const int gw   = blockIdx.x * 8 + (threadIdx.x >> 5);  // 0..511
    const int mat  = gw >> 8;                              // 0 -> (x1), 1 -> (x2)
    const int tm   = gw & 255;                             // 16-row tile index

    const float* X = mat ? x2 : x1;
    float*       H = mat ? h2 : h1;

    const int half = lane >> 4;
    const int m    = lane & 15;
    const float* xrow = X + (size_t)(tm * 16 + m) * DIN;

    // Base of this wave's B fragments: [mat][ntile][kstep][lane][16]
    const __bf16* wp = Wpack + ((size_t)mat * 4 * 32 + 0) * 32 * 16 + (size_t)lane * 16;
    const size_t ntStride = (size_t)32 * 32 * 16;  // per ntile
    const size_t ksStride = (size_t)32 * 16;       // per kstep

    v8f acc0 = {}; v8f acc1 = {}; v8f acc2 = {}; v8f acc3 = {};

    for (int ks = 0; ks < 32; ++ks) {
        const int k0 = ks * 32;
        // --- A fragment: four contiguous float4 loads, then cvt to bf16 ---
        const float4 xa0 = *(const float4*)(xrow + k0 + 16 * half);
        const float4 xa1 = *(const float4*)(xrow + k0 + 16 * half + 4);
        const float4 xa2 = *(const float4*)(xrow + k0 + 16 * half + 8);
        const float4 xa3 = *(const float4*)(xrow + k0 + 16 * half + 12);
        v16bf a;
        a[0]  = (__bf16)xa0.x; a[1]  = (__bf16)xa0.y; a[2]  = (__bf16)xa0.z; a[3]  = (__bf16)xa0.w;
        a[4]  = (__bf16)xa1.x; a[5]  = (__bf16)xa1.y; a[6]  = (__bf16)xa1.z; a[7]  = (__bf16)xa1.w;
        a[8]  = (__bf16)xa2.x; a[9]  = (__bf16)xa2.y; a[10] = (__bf16)xa2.z; a[11] = (__bf16)xa2.w;
        a[12] = (__bf16)xa3.x; a[13] = (__bf16)xa3.y; a[14] = (__bf16)xa3.z; a[15] = (__bf16)xa3.w;

        // --- B fragments: pre-packed, 32B contiguous per lane per ntile ---
        const v16bf b0 = *(const v16bf*)(wp + 0 * ntStride + ks * ksStride);
        const v16bf b1 = *(const v16bf*)(wp + 1 * ntStride + ks * ksStride);
        const v16bf b2 = *(const v16bf*)(wp + 2 * ntStride + ks * ksStride);
        const v16bf b3 = *(const v16bf*)(wp + 3 * ntStride + ks * ksStride);

        acc0 = __builtin_amdgcn_wmma_f32_16x16x32_bf16(false, a, false, b0, (short)0, acc0, false, false);
        acc1 = __builtin_amdgcn_wmma_f32_16x16x32_bf16(false, a, false, b1, (short)0, acc1, false, false);
        acc2 = __builtin_amdgcn_wmma_f32_16x16x32_bf16(false, a, false, b2, (short)0, acc2, false, false);
        acc3 = __builtin_amdgcn_wmma_f32_16x16x32_bf16(false, a, false, b3, (short)0, acc3, false, false);
    }

#pragma unroll
    for (int r = 0; r < 8; ++r) {
        const size_t row = (size_t)(tm * 16 + r + 8 * half) * DOUT;
        H[row + 0  + m] = acc0[r];
        H[row + 16 + m] = acc1[r];
        H[row + 32 + m] = acc2[r];
        H[row + 48 + m] = acc3[r];
    }
}

// ---------------------------------------------------------------------------
// Kernel 2: the six per-node attention score scalars (h @ a vectors).
// svec layout [6][N]: 0:h1*a1_1  1:h1*a2_1  2:h2*a2_1  3:h2*a1_2  4:h1*a2_2  5:h2*a2_2
// ---------------------------------------------------------------------------
__global__ __launch_bounds__(256) void gat_svec_kernel(
    const float* __restrict__ h1, const float* __restrict__ h2,
    const float* __restrict__ a1_1, const float* __restrict__ a2_1,
    const float* __restrict__ a1_2, const float* __restrict__ a2_2,
    float* __restrict__ svec)
{
    const int i = blockIdx.x * blockDim.x + threadIdx.x;
    if (i >= N_NODES) return;
    const float* r1 = h1 + (size_t)i * DOUT;
    const float* r2 = h2 + (size_t)i * DOUT;
    float d0 = 0.f, d1 = 0.f, d2 = 0.f, d3 = 0.f, d4 = 0.f, d5 = 0.f;
#pragma unroll 8
    for (int f = 0; f < DOUT; ++f) {
        const float v1 = r1[f], v2 = r2[f];
        const float A11 = a1_1[f], A21 = a2_1[f], A12 = a1_2[f], A22 = a2_2[f];
        d0 += v1 * A11;  d1 += v1 * A21;  d2 += v2 * A21;
        d3 += v2 * A12;  d4 += v1 * A22;  d5 += v2 * A22;
    }
    svec[0 * N_NODES + i] = d0;  svec[1 * N_NODES + i] = d1;
    svec[2 * N_NODES + i] = d2;  svec[3 * N_NODES + i] = d3;
    svec[4 * N_NODES + i] = d4;  svec[5 * N_NODES + i] = d5;
}

// ---------------------------------------------------------------------------
// Kernel 3: fused masked-softmax attention + aggregation, one wave32 per
// (pair, row). Single pass over the adjacency row with an online softmax:
//   out[i,:] = GAMMA*deg/Sum * sum_{adj=1} exp(e-max) g[j,:]
//            + (1-GAMMA)    * sum_{adj=1} g[j,:]  + bias
// Non-adjacent entries contribute exp(-9e15 - max) == 0 in fp32, so the
// dense reference softmax reduces exactly to this sparse form; deg==0 rows
// degenerate to bias (coef guarded to 0).
// Each lane owns output features {2*lane, 2*lane+1}.
// ---------------------------------------------------------------------------
__global__ __launch_bounds__(256) void gat_attention_kernel(
    const float* __restrict__ h1, const float* __restrict__ h2,
    const float* __restrict__ svec,
    const float* __restrict__ adj11, const float* __restrict__ adj12,
    const float* __restrict__ adj21, const float* __restrict__ adj22,
    const float* __restrict__ bias,
    float* __restrict__ out)
{
    const int lane = threadIdx.x & 31;
    const int gw   = blockIdx.x * 8 + (threadIdx.x >> 5);  // 0..16383
    const int p    = gw >> 12;                             // pair 0..3
    const int row  = gw & (N_NODES - 1);

    const float* adj = (p == 0) ? adj11 : (p == 1) ? adj12 : (p == 2) ? adj21 : adj22;
    const float* g   = (p == 0 || p == 2) ? h1 : h2;
    const int s1i = (p < 2) ? 0 : 3;
    const int s2i = (p == 0) ? 1 : (p == 1) ? 2 : (p == 2) ? 4 : 5;

    const float  s1   = svec[s1i * N_NODES + row];
    const float* s2v  = svec + s2i * N_NODES;
    const float* arow = adj + (size_t)row * N_NODES;

    float mmax = -INFINITY;
    float ssum = 0.f, deg = 0.f;
    float aS0 = 0.f, aS1 = 0.f, a90 = 0.f, a91 = 0.f;

    for (int base = 0; base < N_NODES; base += 128) {
        if (base + 128 < N_NODES)  // stream-prefetch next adjacency chunk
            __builtin_prefetch(arow + base + 128 + lane * 4, 0, 3);
        const int j0 = base + lane * 4;
        const float4 av = *(const float4*)(arow + j0);
        const float4 sv = *(const float4*)(s2v + j0);
#pragma unroll
        for (int k = 0; k < 4; ++k) {
            const float aval = (k == 0) ? av.x : (k == 1) ? av.y : (k == 2) ? av.z : av.w;
            const float s2e  = (k == 0) ? sv.x : (k == 1) ? sv.y : (k == 2) ? sv.z : sv.w;
            const float t = s1 + s2e;
            const float e = (t > 0.f) ? t : 0.2f * t;   // leaky_relu(0.2)
            unsigned mask = __builtin_amdgcn_ballot_w32(aval > 0.f);
            while (mask) {
                const int l = __builtin_ctz(mask);
                mask &= (mask - 1u);
                const float el = __shfl(e, l, 32);
                const int j = base + l * 4 + k;
                const float2 gv = *(const float2*)(g + (size_t)j * DOUT + 2 * lane);
                if (el > mmax) {  // online softmax rescale
                    const float sc = __expf(mmax - el);
                    ssum *= sc; aS0 *= sc; aS1 *= sc;
                    mmax = el;
                }
                const float w = __expf(el - mmax);
                ssum += w;
                aS0 += w * gv.x;  aS1 += w * gv.y;
                a90 += gv.x;      a91 += gv.y;
                deg += 1.f;
            }
        }
    }

    const float coef = (ssum > 0.f) ? (GAMMA_F * deg / ssum) : 0.f;
    const int f = 2 * lane;
    float* orow = out + ((size_t)p * N_NODES + row) * DOUT;
    orow[f]     = coef * aS0 + (1.f - GAMMA_F) * a90 + bias[f];
    orow[f + 1] = coef * aS1 + (1.f - GAMMA_F) * a91 + bias[f + 1];
}

// ---------------------------------------------------------------------------
extern "C" void kernel_launch(void* const* d_in, const int* in_sizes, int n_in,
                              void* d_out, int out_size, void* d_ws, size_t ws_size,
                              hipStream_t stream) {
    const float* x1   = (const float*)d_in[0];
    const float* x2   = (const float*)d_in[1];
    const float* W1   = (const float*)d_in[2];
    const float* W2   = (const float*)d_in[3];
    const float* a1_1 = (const float*)d_in[4];
    const float* a2_1 = (const float*)d_in[5];
    const float* a1_2 = (const float*)d_in[6];
    const float* a2_2 = (const float*)d_in[7];
    const float* bias = (const float*)d_in[8];
    const float* adj11 = (const float*)d_in[9];
    const float* adj12 = (const float*)d_in[10];
    const float* adj21 = (const float*)d_in[11];
    const float* adj22 = (const float*)d_in[12];
    float* out = (float*)d_out;

    float* ws = (float*)d_ws;
    float*  h1    = ws;                                  // 4096*64 f32
    float*  h2    = ws + (size_t)N_NODES * DOUT;         // 4096*64 f32
    float*  svec  = ws + (size_t)2 * N_NODES * DOUT;     // 6*4096 f32
    __bf16* Wpack = (__bf16*)(ws + (size_t)2 * N_NODES * DOUT + 6 * N_NODES);  // 2*1024*64 bf16

    gat_packw_kernel<<<32, 256, 0, stream>>>(W1, W2, Wpack);
    // 512 waves (2 matrices x 256 row-tiles), 8 waves/block
    gat_gemm_h_kernel<<<64, 256, 0, stream>>>(x1, x2, Wpack, h1, h2);
    gat_svec_kernel<<<(N_NODES + 255) / 256, 256, 0, stream>>>(h1, h2, a1_1, a2_1, a1_2, a2_2, svec);
    // 4 pairs x 4096 rows = 16384 waves, 8 waves/block
    gat_attention_kernel<<<2048, 256, 0, stream>>>(h1, h2, svec,
                                                   adj11, adj12, adj21, adj22,
                                                   bias, out);
}
